// Transformer_39178691674311
// MI455X (gfx1250) — compile-verified, matrix operand order checked by reading, not run on hardware
//
#include <hip/hip_runtime.h>
#include <hip/hip_bf16.h>

// ---------------------------------------------------------------------------
// CDNA5 (gfx1250) bf16-WMMA transformer forward.
//   D = A * B + C via v_wmma_f32_16x16x32_bf16, wave32.
//   Double-buffered LDS pipeline, b128 global staging, global_prefetch,
//   native v_cvt bf16 conversion.
// ---------------------------------------------------------------------------

typedef __attribute__((ext_vector_type(16))) __bf16 v16bf;
typedef __attribute__((ext_vector_type(2)))  __bf16 v2bf;
typedef __attribute__((ext_vector_type(8)))  float  v8f;

union FragU { v16bf v; unsigned int u[8]; };

__device__ __forceinline__ unsigned short f2bf(float f) {
    __bf16 h = (__bf16)f;                       // native cvt, RNE
    return __builtin_bit_cast(unsigned short, h);
}
__device__ __forceinline__ unsigned int pk2(float lo, float hi) {
    v2bf t; t[0] = (__bf16)lo; t[1] = (__bf16)hi;   // v_cvt_pk_bf16_f32
    return __builtin_bit_cast(unsigned int, t);
}

// Generic batched GEMM: C[m,n] = scale * sum_k A[m,k]*B(k,n)  (+bias)(+src)
//  A: fp32 row-major [M,K], per-batch offset zb*sA1 + zh*sA2
//  B: fp32; bTrans==1 -> stored [N,K] row-major (torch Linear [out,in]),
//           bTrans==0 -> stored [K,N] row-major
//  bias: fp32, added as bias[(m % bias_rows)*bias_ld + n] (nullable)
//  src : fp32 residual, same indexing as C (nullable)
// Workgroup: 256 threads (8 waves), 128x128 tile, BK=32, double-buffered LDS.
__global__ __launch_bounds__(256)
void gemm_bf16_wmma(const float* __restrict__ A, int lda, long long sA1, long long sA2,
                    const float* __restrict__ B, int ldb, long long sB1, long long sB2, int bTrans,
                    const float* __restrict__ bias, int bias_rows, int bias_ld,
                    const float* __restrict__ src,
                    float* __restrict__ C, int ldc, long long sC1, long long sC2,
                    int M, int N, int K, int Hs, float scale)
{
    __shared__ unsigned short As[2][128 * 36];   // [m][k], k-contiguous, stride 36
    __shared__ unsigned short Bs[2][128 * 36];   // [n][k], k-contiguous, stride 36

    const int tid  = threadIdx.x;
    const int wave = tid >> 5;
    const int lane = tid & 31;
    const int wr   = wave >> 1;      // 0..3  (32-row strip)
    const int wc   = wave & 1;       // 0..1  (64-col strip)
    const int half = lane >> 4;      // 0/1
    const int l16  = lane & 15;

    const int z  = blockIdx.z;
    const int zb = z / Hs;
    const int zh = z % Hs;
    const long long offA = (long long)zb * sA1 + (long long)zh * sA2;
    const long long offB = (long long)zb * sB1 + (long long)zh * sB2;
    const long long offC = (long long)zb * sC1 + (long long)zh * sC2;

    const int mblk = blockIdx.y * 128;
    const int nblk = blockIdx.x * 128;

    const float* Abase = A + offA + (long long)mblk * lda;
    const float* BbaseT = B + offB + (long long)nblk * ldb;   // bTrans path
    const float* BbaseN = B + offB + nblk;                    // normal path

    // Per-thread staging coordinates (4 float4 quads each for A and B).
    int am[4], ak[4], b0[4], b1[4];
#pragma unroll
    for (int i = 0; i < 4; ++i) {
        const int q = tid + (i << 8);            // 0..1023 quad index
        am[i] = q >> 3;  ak[i] = q & 7;          // A: 128 rows x 8 quads
        if (bTrans) { b0[i] = q >> 3; b1[i] = q & 7;  }   // [n][k-quad]
        else        { b0[i] = q >> 5; b1[i] = q & 31; }   // [k][n-quad]
    }

    v8f acc[2][4];
#pragma unroll
    for (int r = 0; r < 2; ++r)
#pragma unroll
        for (int c = 0; c < 4; ++c)
#pragma unroll
            for (int i = 0; i < 8; ++i) acc[r][c][i] = 0.0f;

    const int ksteps = K >> 5;
    float4 ra[4], rb[4];

    // ---- staging helpers ----
    auto loadTiles = [&](int kt) {
        const float* Ap = Abase + (kt << 5);
#pragma unroll
        for (int i = 0; i < 4; ++i)
            ra[i] = *(const float4*)(Ap + (long long)am[i] * lda + (ak[i] << 2));
        if (bTrans) {
            const float* Bp = BbaseT + (kt << 5);
#pragma unroll
            for (int i = 0; i < 4; ++i)
                rb[i] = *(const float4*)(Bp + (long long)b0[i] * ldb + (b1[i] << 2));
        } else {
            const float* Bp = BbaseN + (long long)(kt << 5) * ldb;
#pragma unroll
            for (int i = 0; i < 4; ++i)
                rb[i] = *(const float4*)(Bp + (long long)b0[i] * ldb + (b1[i] << 2));
        }
    };
    auto storeTiles = [&](int buf) {
#pragma unroll
        for (int i = 0; i < 4; ++i) {
            uint2 w; w.x = pk2(ra[i].x, ra[i].y); w.y = pk2(ra[i].z, ra[i].w);
            *(uint2*)&As[buf][am[i] * 36 + (ak[i] << 2)] = w;
        }
        if (bTrans) {
#pragma unroll
            for (int i = 0; i < 4; ++i) {
                uint2 w; w.x = pk2(rb[i].x, rb[i].y); w.y = pk2(rb[i].z, rb[i].w);
                *(uint2*)&Bs[buf][b0[i] * 36 + (b1[i] << 2)] = w;
            }
        } else {
#pragma unroll
            for (int i = 0; i < 4; ++i) {      // transpose into [n][k]
                const int kk = b0[i], n0 = b1[i] << 2;
                Bs[buf][(n0 + 0) * 36 + kk] = f2bf(rb[i].x);
                Bs[buf][(n0 + 1) * 36 + kk] = f2bf(rb[i].y);
                Bs[buf][(n0 + 2) * 36 + kk] = f2bf(rb[i].z);
                Bs[buf][(n0 + 3) * 36 + kk] = f2bf(rb[i].w);
            }
        }
    };

    // ---- pipeline prologue ----
    loadTiles(0);
    storeTiles(0);
    __syncthreads();

    for (int kt = 0; kt < ksteps; ++kt) {
        const int cur = kt & 1;
        const bool more = (kt + 1) < ksteps;
        if (more) loadTiles(kt + 1);
        if (kt + 2 < ksteps) {   // keep L2 -> WGP path warm
            __builtin_prefetch(Abase + ((kt + 2) << 5) + (long long)am[0] * lda, 0, 3);
            __builtin_prefetch((bTrans ? BbaseT + ((kt + 2) << 5)
                                       : BbaseN + (long long)((kt + 2) << 5) * ldb)
                               + (long long)b0[0] * ldb, 0, 3);
        }

        // ---- fragments per documented 16-bit A/B VGPR layout ----
        v16bf afrag[2], bfrag[4];
#pragma unroll
        for (int r = 0; r < 2; ++r) {
            FragU fu;
            const int m = wr * 32 + r * 16 + l16;
#pragma unroll
            for (int v = 0; v < 8; ++v) {
                int k0 = ((v & 4) << 2) + (half << 3) + ((v & 3) << 1);
                fu.u[v] = *(const unsigned int*)&As[cur][m * 36 + k0];
            }
            afrag[r] = fu.v;
        }
#pragma unroll
        for (int c = 0; c < 4; ++c) {
            FragU fu;
            const int n = wc * 64 + c * 16 + l16;
#pragma unroll
            for (int v = 0; v < 8; ++v) {
                int k0 = ((v & 4) << 2) + (half << 3) + ((v & 3) << 1);
                fu.u[v] = *(const unsigned int*)&Bs[cur][n * 36 + k0];
            }
            bfrag[c] = fu.v;
        }

#pragma unroll
        for (int r = 0; r < 2; ++r)
#pragma unroll
            for (int c = 0; c < 4; ++c)
                acc[r][c] = __builtin_amdgcn_wmma_f32_16x16x32_bf16(
                    false, afrag[r], false, bfrag[c],
                    (short)0, acc[r][c], false, false);

        if (more) storeTiles(cur ^ 1);
        __syncthreads();
    }

    // ---- epilogue: scale, bias (row-modulo), residual, store fp32 ----
#pragma unroll
    for (int r = 0; r < 2; ++r) {
#pragma unroll
        for (int c = 0; c < 4; ++c) {
            const int n = nblk + wc * 64 + c * 16 + l16;
#pragma unroll
            for (int v = 0; v < 8; ++v) {
                const int m = mblk + wr * 32 + r * 16 + half * 8 + v;
                float val = acc[r][c][v] * scale;
                if (bias) val += bias[(long long)(m % bias_rows) * bias_ld + n];
                const long long ci = offC + (long long)m * ldc + n;
                if (src) val += src[ci];
                C[ci] = val;
            }
        }
    }
}

// In-place softmax over rows of length 1024. One block (256 thr) per row.
__global__ __launch_bounds__(256)
void softmax1024(float* __restrict__ p)
{
    __shared__ float red[256];
    const int tid = threadIdx.x;
    float* row = p + (size_t)blockIdx.x * 1024;

    float v[4];
    float m = -3.402823466e38f;
#pragma unroll
    for (int i = 0; i < 4; ++i) { v[i] = row[tid + (i << 8)]; m = fmaxf(m, v[i]); }
    red[tid] = m; __syncthreads();
    for (int s = 128; s > 0; s >>= 1) {
        if (tid < s) red[tid] = fmaxf(red[tid], red[tid + s]);
        __syncthreads();
    }
    m = red[0]; __syncthreads();

    float sum = 0.0f;
#pragma unroll
    for (int i = 0; i < 4; ++i) { v[i] = __expf(v[i] - m); sum += v[i]; }
    red[tid] = sum; __syncthreads();
    for (int s = 128; s > 0; s >>= 1) {
        if (tid < s) red[tid] += red[tid + s];
        __syncthreads();
    }
    const float inv = 1.0f / red[0];
#pragma unroll
    for (int i = 0; i < 4; ++i) row[tid + (i << 8)] = v[i] * inv;
}

// wo[l]: [H=8][N=1024][P=128] -> woT: [N=1024][H*P=1024]
__global__ __launch_bounds__(256)
void transpose_wo(const float* __restrict__ wo_l, float* __restrict__ woT)
{
    const int idx = blockIdx.x * 256 + threadIdx.x;      // 0 .. 2^20-1
    const int h   = idx >> 17;
    const int rem = idx & 131071;
    const int n   = rem >> 7;
    const int p   = rem & 127;
    woT[(size_t)n * 1024 + h * 128 + p] = wo_l[idx];
}

// ---------------------------------------------------------------------------

static void launch_gemm(hipStream_t stream,
                        const float* A, int lda, long long sA1, long long sA2,
                        const float* B, int ldb, long long sB1, long long sB2, int bTrans,
                        const float* bias, int bias_rows, int bias_ld,
                        const float* src,
                        float* C, int ldc, long long sC1, long long sC2,
                        int M, int N, int K, int batches, int Hs, float scale)
{
    dim3 grid(N / 128, M / 128, batches);
    gemm_bf16_wmma<<<grid, dim3(256), 0, stream>>>(
        A, lda, sA1, sA2, B, ldb, sB1, sB2, bTrans,
        bias, bias_rows, bias_ld, src, C, ldc, sC1, sC2, M, N, K, Hs, scale);
}

extern "C" void kernel_launch(void* const* d_in, const int* in_sizes, int n_in,
                              void* d_out, int out_size, void* d_ws, size_t ws_size,
                              hipStream_t stream)
{
    (void)in_sizes; (void)n_in; (void)out_size; (void)ws_size;

    constexpr int Bb = 4, S = 1024, Nst = 1024, V = 512, H = 8, P = 128, D = 4;
    constexpr int BS = Bb * S;                       // 4096
    constexpr float rscale = 0.08838834764831845f;   // 1/sqrt(128)

    const float* tokens   = (const float*)d_in[0];   // [B,S,V]
    const float* att_mask = (const float*)d_in[1];   // [S,S]
    const float* embed_w  = (const float*)d_in[2];   // [N,V]
    const float* unembed  = (const float*)d_in[3];   // [V,N]
    const float* pos_enc  = (const float*)d_in[4];   // [S,N]
    const float* wq       = (const float*)d_in[5];   // [D,H,P,N]
    const float* wk       = (const float*)d_in[6];
    const float* wv       = (const float*)d_in[7];
    const float* wo       = (const float*)d_in[8];   // [D,H,N,P]

    float* out = (float*)d_out;
    float* logits = out;                                       // [B,S,V]
    float* atts   = out + (size_t)Bb * S * V;                  // [D,B,H,S,S]
    const size_t att_l_sz = (size_t)Bb * H * S * S;
    float* streams = atts + (size_t)D * att_l_sz;              // [D+1,B,S,N]
    const size_t stream_sz = (size_t)Bb * S * Nst;

    float* ws   = (float*)d_ws;
    float* qb   = ws;                                          // [BS, H*P]
    float* kb   = qb + (size_t)BS * Nst;
    float* vb   = kb + (size_t)BS * Nst;
    float* comb = vb + (size_t)BS * Nst;                       // [BS, H*P]
    float* woT  = comb + (size_t)BS * Nst;                     // [N, H*P]

    // 1) embed: stream0 = tokens @ embed_w^T + pos_enc
    launch_gemm(stream,
                tokens, V, 0, 0,
                embed_w, V, 0, 0, /*bTrans=*/1,
                pos_enc, S, Nst,
                nullptr,
                streams, Nst, 0, 0,
                BS, Nst, V, 1, 1, 1.0f);

    for (int l = 0; l < D; ++l) {
        const float* stream_l = streams + (size_t)l * stream_sz;
        float* stream_n = streams + (size_t)(l + 1) * stream_sz;
        float* att_l = atts + (size_t)l * att_l_sz;
        const size_t wOff = (size_t)l * H * P * Nst;           // 1M floats/layer

        // 2) q,k,v = stream @ w{q,k,v}[l]^T   -> [BS, H*P]
        launch_gemm(stream, stream_l, Nst, 0, 0, wq + wOff, Nst, 0, 0, 1,
                    nullptr, 1, 0, nullptr, qb, Nst, 0, 0, BS, Nst, Nst, 1, 1, 1.0f);
        launch_gemm(stream, stream_l, Nst, 0, 0, wk + wOff, Nst, 0, 0, 1,
                    nullptr, 1, 0, nullptr, kb, Nst, 0, 0, BS, Nst, Nst, 1, 1, 1.0f);
        launch_gemm(stream, stream_l, Nst, 0, 0, wv + wOff, Nst, 0, 0, 1,
                    nullptr, 1, 0, nullptr, vb, Nst, 0, 0, BS, Nst, Nst, 1, 1, 1.0f);

        // 3) logits = (q @ k^T) / sqrt(P) + mask, batched over (b,h)
        launch_gemm(stream,
                    qb, Nst, (long long)S * Nst, P,
                    kb, Nst, (long long)S * Nst, P, /*bTrans=*/1,
                    att_mask, S, S,
                    nullptr,
                    att_l, S, (long long)H * S * S, (long long)S * S,
                    S, S, P, Bb * H, H, rscale);

        // 4) softmax in place on att_l
        softmax1024<<<dim3(Bb * H * S), dim3(256), 0, stream>>>(att_l);

        // 5) comb = att @ v, batched over (b,h)  -> packed [BS, H*P]
        launch_gemm(stream,
                    att_l, S, (long long)H * S * S, (long long)S * S,
                    vb, Nst, (long long)S * Nst, P, /*bTrans=*/0,
                    nullptr, 1, 0,
                    nullptr,
                    comb, Nst, (long long)S * Nst, P,
                    S, P, S, Bb * H, H, 1.0f);

        // 6) wo[l] -> woT, then stream_{l+1} = stream_l + comb @ woT^T
        transpose_wo<<<dim3((H * Nst * P) / 256), dim3(256), 0, stream>>>(
            wo + wOff, woT);
        launch_gemm(stream,
                    comb, Nst, 0, 0,
                    woT, Nst, 0, 0, /*bTrans=*/1,
                    nullptr, 1, 0,
                    stream_l,
                    stream_n, Nst, 0, 0,
                    BS, Nst, Nst, 1, 1, 1.0f);
    }

    // 7) logits = stream_D @ unembed^T
    launch_gemm(stream,
                streams + (size_t)D * stream_sz, Nst, 0, 0,
                unembed, Nst, 0, 0, /*bTrans=*/1,
                nullptr, 1, 0,
                nullptr,
                logits, V, 0, 0,
                BS, V, Nst, 1, 1, 1.0f);
}
